// MixAttention_34651796144665
// MI455X (gfx1250) — matrix operand order repaired and checked by reading, not compiled, not run on hardware
//
#include <hip/hip_runtime.h>
#include <hip/hip_bf16.h>

typedef __bf16 bf16;
typedef __attribute__((ext_vector_type(16))) __bf16 v16bf;
typedef __attribute__((ext_vector_type(8)))  __bf16 v8bf;
typedef __attribute__((ext_vector_type(4)))  __bf16 v4bf;
typedef __attribute__((ext_vector_type(8)))  float  v8f;
typedef __attribute__((ext_vector_type(4)))  float  f32x4;

// explicit global address space so loads select global_load (saddr+voffset), not flat
#define AS_G __attribute__((address_space(1)))
typedef AS_G const bf16*   gc_bf16;
typedef AS_G const float*  gc_f32;
typedef AS_G float*        g_f32;
typedef AS_G const int*    gc_i32;
typedef AS_G const v16bf*  gcp_v16bf;
typedef AS_G const f32x4*  gcp_f32x4;

#define NTOK   49
#define NPAD   64
#define DMODEL 256
#define HD     32
#define NHEAD  8
#define BATCH  4096
#define NWIN   64
#define QSCALE 0.17677669529663687f

// LDS layout (element offsets into the dynamic bf16 array)
#define O_SX   0                       // [64][256]
#define O_SY   (O_SX + NPAD * DMODEL)  // [64][256]
#define O_SOX  (O_SY + NPAD * DMODEL)  // [64][256]
#define O_SOY  (O_SOX + NPAD * DMODEL) // [64][256]
#define O_SA   (O_SOY + NPAD * DMODEL) // 8 x [64][32]
#define O_SP   (O_SA + NHEAD * NPAD * HD) // 8 x [64][64]
#define LDS_ELEMS (O_SP + NHEAD * NPAD * NPAD)

// ---------------- fragment helpers (CDNA5 WMMA 16x16x32 bf16 layouts) ----------------

// A-matrix 16x32 bf16: lane L: row = L%16; lanes 0-15 hold K=0..15,
// lanes 16-31 hold K=16..31 (matches B-frag order). Source: LDS, 32-bit offsets.
__device__ __forceinline__ v16bf lds_load_af(int base, int stride, int row, int col, int lane) {
    extern __shared__ bf16 smem[];
    int m   = lane & 15;
    int kb  = (lane < 16) ? 0 : 16;
    int off = base + (row + m) * stride + col + kb;
    v8bf lo = *(const v8bf*)&smem[off];
    v8bf hi = *(const v8bf*)&smem[off + 8];
    return __builtin_shufflevector(lo, hi, 0,1,2,3,4,5,6,7,8,9,10,11,12,13,14,15);
}

// B-matrix 32x16 bf16 from B^T in LDS (row-major [Ncol][K]): lane L: col = L%16;
// lanes 0-15 hold K=0..15, lanes 16-31 hold K=16..31 -> 16 contiguous bf16.
__device__ __forceinline__ v16bf lds_load_bfrag(int base, int stride, int colrow, int kbase, int lane) {
    extern __shared__ bf16 smem[];
    int n   = lane & 15;
    int kb  = (lane < 16) ? 0 : 16;
    int off = base + (colrow + n) * stride + kbase + kb;
    return *(const v16bf*)&smem[off];
}

// B-fragment from a transposed weight matrix in global memory (uniform base + 32-bit offset)
__device__ __forceinline__ v16bf glb_load_bfrag(gc_bf16 baseT, int stride,
                                                int colrow, int kbase, int lane) {
    int n   = lane & 15;
    int kb  = (lane < 16) ? 0 : 16;
    int off = (colrow + n) * stride + kbase + kb;
    return *(gcp_v16bf)(baseT + off);
}

__device__ __forceinline__ v8f wmma_bf16(v16bf a, v16bf b, v8f c) {
    return __builtin_amdgcn_wmma_f32_16x16x32_bf16(false, a, false, b, (short)0, c, false, false);
}

__device__ __forceinline__ v8f zero8() {
    v8f z = {0.f,0.f,0.f,0.f,0.f,0.f,0.f,0.f};
    return z;
}

// C = A(64x256 LDS bf16 at aBase) @ B(256x32), B given transposed in global (WT: [32][256]).
// Adds per-column bias, scales, writes bf16 to LDS at outBase: row-major [64][32] or V^T [32][64].
__device__ void gemm64x32(int aBase, gc_bf16 WT, gc_f32 bias,
                          float scale, int outBase, bool transposed, int lane) {
    extern __shared__ bf16 smem[];
    v8f acc[4][2];
    #pragma unroll
    for (int mi = 0; mi < 4; ++mi)
        #pragma unroll
        for (int nj = 0; nj < 2; ++nj) acc[mi][nj] = zero8();

    #pragma unroll
    for (int ks = 0; ks < 8; ++ks) {
        v16bf a[4];
        #pragma unroll
        for (int mi = 0; mi < 4; ++mi) a[mi] = lds_load_af(aBase, DMODEL, mi * 16, ks * 32, lane);
        #pragma unroll
        for (int nj = 0; nj < 2; ++nj) {
            v16bf bb = glb_load_bfrag(WT, DMODEL, nj * 16, ks * 32, lane);
            #pragma unroll
            for (int mi = 0; mi < 4; ++mi) acc[mi][nj] = wmma_bf16(a[mi], bb, acc[mi][nj]);
        }
    }
    int n  = lane & 15;
    int mo = (lane < 16) ? 0 : 8;
    #pragma unroll
    for (int nj = 0; nj < 2; ++nj) {
        float bi = bias[nj * 16 + n];
        #pragma unroll
        for (int mi = 0; mi < 4; ++mi) {
            if (!transposed) {
                #pragma unroll
                for (int v = 0; v < 8; ++v)
                    smem[outBase + (mi * 16 + v + mo) * HD + nj * 16 + n] =
                        (bf16)((acc[mi][nj][v] + bi) * scale);
            } else {
                v8bf pk;
                #pragma unroll
                for (int v = 0; v < 8; ++v) pk[v] = (bf16)((acc[mi][nj][v] + bi) * scale);
                *(v8bf*)&smem[outBase + (nj * 16 + n) * NPAD + mi * 16 + mo] = pk;  // V^T: [32][64]
            }
        }
    }
}

__shared__ float sM[NPAD];

// One attention stream for one head (one wave): softmax((Aq@Wq)*(Akv@Wk)^T + bias) @ (Akv@Wv)
__device__ void attn_phase(int aqBase,  gc_bf16 WqT, gc_f32 qb,
                           int akvBase, gc_bf16 WkT, gc_f32 kb_,
                           gc_bf16 WvT, gc_f32 vb_,
                           int saBase, int spBase, gc_f32 biasB,
                           int outBase, int lane) {
    extern __shared__ bf16 smem[];

    // Q (scaled) -> SA, then A-fragments into registers
    gemm64x32(aqBase, WqT, qb, QSCALE, saBase, false, lane);
    v16bf qf[4];
    #pragma unroll
    for (int mi = 0; mi < 4; ++mi) qf[mi] = lds_load_af(saBase, HD, mi * 16, 0, lane);

    // K -> SA (row-major: B-frag of K^T is a contiguous K row)
    gemm64x32(akvBase, WkT, kb_, 1.0f, saBase, false, lane);

    // attn = Q @ K^T  (64x64, K-dim = 32 in a single WMMA step)
    v8f at[4][4];
    #pragma unroll
    for (int mi = 0; mi < 4; ++mi)
        #pragma unroll
        for (int nj = 0; nj < 4; ++nj) at[mi][nj] = zero8();
    #pragma unroll
    for (int nj = 0; nj < 4; ++nj) {
        v16bf kf = lds_load_bfrag(saBase, HD, nj * 16, 0, lane);
        #pragma unroll
        for (int mi = 0; mi < 4; ++mi) at[mi][nj] = wmma_bf16(qf[mi], kf, at[mi][nj]);
    }

    int n  = lane & 15;
    int mo = (lane < 16) ? 0 : 8;

    // hoisted nuclei-mask reads (sM[col] per nj, sM[row] per (mi,v))
    float mc[4];
    #pragma unroll
    for (int nj = 0; nj < 4; ++nj) mc[nj] = sM[nj * 16 + n];

    // biases: precomputed (window-mask + rpb) + nuclei(m_q + m_k); mask invalid cols
    #pragma unroll
    for (int mi = 0; mi < 4; ++mi) {
        float mr[8];
        #pragma unroll
        for (int v = 0; v < 8; ++v) mr[v] = sM[mi * 16 + v + mo];
        #pragma unroll
        for (int nj = 0; nj < 4; ++nj)
            #pragma unroll
            for (int v = 0; v < 8; ++v) {
                int row = mi * 16 + v + mo;
                int col = nj * 16 + n;
                float val;
                if (col < NTOK) {
                    val = at[mi][nj][v] + mr[v] + mc[nj];
                    if (row < NTOK) val += biasB[row * NTOK + col];
                } else {
                    val = -1.0e30f;
                }
                at[mi][nj][v] = val;
            }
    }

    // row softmax: row lives in one 16-lane half (C layout) -> xor-shuffle 1,2,4,8
    #pragma unroll
    for (int mi = 0; mi < 4; ++mi)
        #pragma unroll
        for (int v = 0; v < 8; ++v) {
            float mx = at[mi][0][v];
            mx = fmaxf(mx, at[mi][1][v]);
            mx = fmaxf(mx, at[mi][2][v]);
            mx = fmaxf(mx, at[mi][3][v]);
            #pragma unroll
            for (int off = 1; off < 16; off <<= 1) mx = fmaxf(mx, __shfl_xor(mx, off, 32));
            float s = 0.f;
            #pragma unroll
            for (int nj = 0; nj < 4; ++nj) {
                float e = __expf(at[mi][nj][v] - mx);
                at[mi][nj][v] = e;
                s += e;
            }
            #pragma unroll
            for (int off = 1; off < 16; off <<= 1) s += __shfl_xor(s, off, 32);
            float inv = 1.0f / s;
            #pragma unroll
            for (int nj = 0; nj < 4; ++nj) at[mi][nj][v] *= inv;
        }

    // probabilities -> SP (bf16, [64][64])
    #pragma unroll
    for (int mi = 0; mi < 4; ++mi)
        #pragma unroll
        for (int nj = 0; nj < 4; ++nj)
            #pragma unroll
            for (int v = 0; v < 8; ++v)
                smem[spBase + (mi * 16 + v + mo) * NPAD + nj * 16 + n] = (bf16)at[mi][nj][v];

    // V -> SA transposed ([32][64]) so B-frags of V are contiguous
    gemm64x32(akvBase, WvT, vb_, 1.0f, saBase, true, lane);

    // out = P(64x64) @ V(64x32)
    v8f oc[4][2];
    #pragma unroll
    for (int mi = 0; mi < 4; ++mi)
        #pragma unroll
        for (int nj = 0; nj < 2; ++nj) oc[mi][nj] = zero8();
    #pragma unroll
    for (int ks = 0; ks < 2; ++ks) {
        v16bf pa[4];
        #pragma unroll
        for (int mi = 0; mi < 4; ++mi) pa[mi] = lds_load_af(spBase, NPAD, mi * 16, ks * 32, lane);
        #pragma unroll
        for (int nj = 0; nj < 2; ++nj) {
            v16bf vf = lds_load_bfrag(saBase, NPAD, nj * 16, ks * 32, lane);
            #pragma unroll
            for (int mi = 0; mi < 4; ++mi) oc[mi][nj] = wmma_bf16(pa[mi], vf, oc[mi][nj]);
        }
    }
    #pragma unroll
    for (int mi = 0; mi < 4; ++mi)
        #pragma unroll
        for (int nj = 0; nj < 2; ++nj)
            #pragma unroll
            for (int v = 0; v < 8; ++v)
                smem[outBase + (mi * 16 + v + mo) * DMODEL + nj * 16 + n] = (bf16)oc[mi][nj][v];
}

// final: out(64x256 LDS bf16) @ W(256x256, given transposed, global) + bias -> f32 global (49 rows)
__device__ void final_proj(int soBase, gc_bf16 WT, gc_f32 pb,
                           g_f32 gout, int wave, int lane) {
    v8f acc[4][2];
    #pragma unroll
    for (int mi = 0; mi < 4; ++mi)
        #pragma unroll
        for (int j = 0; j < 2; ++j) acc[mi][j] = zero8();
    int nj0 = wave * 2;
    #pragma unroll
    for (int ks = 0; ks < 8; ++ks) {
        v16bf a[4];
        #pragma unroll
        for (int mi = 0; mi < 4; ++mi) a[mi] = lds_load_af(soBase, DMODEL, mi * 16, ks * 32, lane);
        #pragma unroll
        for (int j = 0; j < 2; ++j) {
            v16bf bb = glb_load_bfrag(WT, DMODEL, (nj0 + j) * 16, ks * 32, lane);
            #pragma unroll
            for (int mi = 0; mi < 4; ++mi) acc[mi][j] = wmma_bf16(a[mi], bb, acc[mi][j]);
        }
    }
    int n  = lane & 15;
    int mo = (lane < 16) ? 0 : 8;
    #pragma unroll
    for (int j = 0; j < 2; ++j) {
        int col = (nj0 + j) * 16 + n;
        float bi = pb[col];
        #pragma unroll
        for (int mi = 0; mi < 4; ++mi)
            #pragma unroll
            for (int v = 0; v < 8; ++v) {
                int row = mi * 16 + v + mo;
                if (row < NTOK) gout[row * DMODEL + col] = acc[mi][j][v] + bi;
            }
    }
}

// ---------------- pre-kernels ----------------

// Wt[o][k] = (bf16) W[k][o]    (W: K x Ncol row-major)
__global__ void convT(const float* __restrict__ W, bf16* __restrict__ Wt, int K, int Ncol) {
    int i = blockIdx.x * 256 + threadIdx.x;
    int total = K * Ncol;
    if (i < total) {
        int o = i / K;
        int k = i - o * K;
        Wt[i] = (bf16)W[k * Ncol + o];
    }
}

// bias[w][h][i][j] = window_mask[w][i][j] + rpb_table[rel_pos_index[i][j]][h]
__global__ void build_bias(const float* __restrict__ mask, const float* __restrict__ rpb,
                           const int* __restrict__ rpi, float* __restrict__ out) {
    int i = blockIdx.x * 256 + threadIdx.x;
    const int total = NWIN * NHEAD * NTOK * NTOK;
    if (i < total) {
        int j = i % NTOK; int t = i / NTOK;
        int ii = t % NTOK; t /= NTOK;
        int h = t % NHEAD; int w = t / NHEAD;
        out[i] = mask[(w * NTOK + ii) * NTOK + j] + rpb[rpi[ii * NTOK + j] * NHEAD + h];
    }
}

// ---------------- main kernel: one block per window, one wave per head ----------------

__global__ __launch_bounds__(256)
void mixattn_kernel(const float* x_, const float* y_, const float* nmask_,
                    const bf16* WkvXT_, const bf16* WkvYT_,
                    const bf16* WqxT_,  const bf16* WqyT_,
                    const bf16* WpT_,   const bf16* WpyT_,
                    const float* qkv_b_,  const float* qkv_y_b_,
                    const float* qx_b_,   const float* qy_b_,
                    const float* p_b_,    const float* py_b_,
                    const float* biasWH_,
                    float* outX_, float* outY_, float* outM_) {
    extern __shared__ bf16 smem[];

    // promote every pointer to the global address space explicitly
    gc_f32  x      = (gc_f32)(unsigned long long)x_;
    gc_f32  y      = (gc_f32)(unsigned long long)y_;
    gc_f32  nmask  = (gc_f32)(unsigned long long)nmask_;
    gc_bf16 WkvXT  = (gc_bf16)(unsigned long long)WkvXT_;
    gc_bf16 WkvYT  = (gc_bf16)(unsigned long long)WkvYT_;
    gc_bf16 WqxT   = (gc_bf16)(unsigned long long)WqxT_;
    gc_bf16 WqyT   = (gc_bf16)(unsigned long long)WqyT_;
    gc_bf16 WpT    = (gc_bf16)(unsigned long long)WpT_;
    gc_bf16 WpyT   = (gc_bf16)(unsigned long long)WpyT_;
    gc_f32  qkv_b  = (gc_f32)(unsigned long long)qkv_b_;
    gc_f32  qkv_y_b= (gc_f32)(unsigned long long)qkv_y_b_;
    gc_f32  qx_b   = (gc_f32)(unsigned long long)qx_b_;
    gc_f32  qy_b   = (gc_f32)(unsigned long long)qy_b_;
    gc_f32  p_b    = (gc_f32)(unsigned long long)p_b_;
    gc_f32  py_b   = (gc_f32)(unsigned long long)py_b_;
    gc_f32  biasWH = (gc_f32)(unsigned long long)biasWH_;
    g_f32   outX   = (g_f32)(unsigned long long)outX_;
    g_f32   outY   = (g_f32)(unsigned long long)outY_;
    g_f32   outM   = (g_f32)(unsigned long long)outM_;

    int b    = blockIdx.x;
    int tid  = threadIdx.x;
    int wave = tid >> 5;
    int lane = tid & 31;

    // stage x[b], y[b] into LDS as zero-padded 64x256 bf16 (vectorized: float4 -> 4x bf16)
    gc_f32 xb = x + (size_t)b * NTOK * DMODEL;
    gc_f32 yb = y + (size_t)b * NTOK * DMODEL;
    #pragma unroll 4
    for (int i4 = tid; i4 < NPAD * DMODEL / 4; i4 += 256) {
        int i = i4 * 4;
        int r = i >> 8;
        int c = i & 255;
        f32x4 xv = {0.f, 0.f, 0.f, 0.f};
        f32x4 yv = {0.f, 0.f, 0.f, 0.f};
        if (r < NTOK) {
            xv = *(gcp_f32x4)(xb + r * DMODEL + c);
            yv = *(gcp_f32x4)(yb + r * DMODEL + c);
        }
        v4bf xo, yo;
        #pragma unroll
        for (int e = 0; e < 4; ++e) { xo[e] = (bf16)xv[e]; yo[e] = (bf16)yv[e]; }
        *(v4bf*)&smem[O_SX + i] = xo;
        *(v4bf*)&smem[O_SY + i] = yo;
    }
    if (tid < NPAD) sM[tid] = (tid < NTOK) ? nmask[b * NTOK + tid] : 0.0f;
    if (tid < NTOK) outM[b * NTOK + tid] = nmask[b * NTOK + tid];
    __syncthreads();

    int h = wave;
    int saBase = O_SA + wave * NPAD * HD;
    int spBase = O_SP + wave * NPAD * NPAD;
    gc_f32 biasB = biasWH + (size_t)((b & (NWIN - 1)) * NHEAD + h) * NTOK * NTOK;

    // attn_x: q_y (from y) vs k,v (from x) -> x-side head output
    attn_phase(O_SY, WqyT + h * HD * DMODEL, qy_b + h * HD,
               O_SX, WkvXT + h * HD * DMODEL, qkv_b + h * HD,
               WkvXT + (DMODEL + h * HD) * DMODEL, qkv_b + DMODEL + h * HD,
               saBase, spBase, biasB, O_SOX + h * HD, lane);
    // attn_y: q (from x) vs k_y,v_y (from y) -> y-side head output
    attn_phase(O_SX, WqxT + h * HD * DMODEL, qx_b + h * HD,
               O_SY, WkvYT + h * HD * DMODEL, qkv_y_b + h * HD,
               WkvYT + (DMODEL + h * HD) * DMODEL, qkv_y_b + DMODEL + h * HD,
               saBase, spBase, biasB, O_SOY + h * HD, lane);
    __syncthreads();

    final_proj(O_SOX, WpT,  p_b,  outX + (size_t)b * NTOK * DMODEL, wave, lane);
    final_proj(O_SOY, WpyT, py_b, outY + (size_t)b * NTOK * DMODEL, wave, lane);
}

// ---------------- host launch ----------------

extern "C" void kernel_launch(void* const* d_in, const int* in_sizes, int n_in,
                              void* d_out, int out_size, void* d_ws, size_t ws_size,
                              hipStream_t stream) {
    const float* x        = (const float*)d_in[0];
    const float* y        = (const float*)d_in[1];
    const float* nmask    = (const float*)d_in[2];
    const float* mask     = (const float*)d_in[3];
    const float* rpb_tab  = (const float*)d_in[4];
    const float* qkv_w    = (const float*)d_in[5];
    const float* qkv_b    = (const float*)d_in[6];
    const float* qkv_y_w  = (const float*)d_in[7];
    const float* qkv_y_b  = (const float*)d_in[8];
    const float* qx_w     = (const float*)d_in[9];
    const float* qx_b     = (const float*)d_in[10];
    const float* qy_w     = (const float*)d_in[11];
    const float* qy_b     = (const float*)d_in[12];
    const float* p_w      = (const float*)d_in[13];
    const float* p_b      = (const float*)d_in[14];
    const float* py_w     = (const float*)d_in[15];
    const float* py_b     = (const float*)d_in[16];
    const int*   rpi      = (const int*)d_in[17];

    // workspace layout (bf16 weights transposed, then f32 fused bias): ~5.7 MB
    bf16* wsb    = (bf16*)d_ws;
    bf16* WkvXT  = wsb;                  // [512][256]
    bf16* WkvYT  = wsb + 512 * 256;      // [512][256]
    bf16* WqxT   = WkvYT + 512 * 256;    // [256][256]
    bf16* WqyT   = WqxT + 256 * 256;
    bf16* WpT    = WqyT + 256 * 256;
    bf16* WpyT   = WpT + 256 * 256;
    float* biasWH = (float*)(WpyT + 256 * 256);  // [64][8][49][49]

    convT<<<(512 * 256 + 255) / 256, 256, 0, stream>>>(qkv_w,   WkvXT, 256, 512);
    convT<<<(512 * 256 + 255) / 256, 256, 0, stream>>>(qkv_y_w, WkvYT, 256, 512);
    convT<<<(256 * 256 + 255) / 256, 256, 0, stream>>>(qx_w, WqxT, 256, 256);
    convT<<<(256 * 256 + 255) / 256, 256, 0, stream>>>(qy_w, WqyT, 256, 256);
    convT<<<(256 * 256 + 255) / 256, 256, 0, stream>>>(p_w,  WpT,  256, 256);
    convT<<<(256 * 256 + 255) / 256, 256, 0, stream>>>(py_w, WpyT, 256, 256);
    build_bias<<<(NWIN * NHEAD * NTOK * NTOK + 255) / 256, 256, 0, stream>>>(mask, rpb_tab, rpi, biasWH);

    float* outX = (float*)d_out;
    float* outY = outX + (size_t)BATCH * NTOK * DMODEL;
    float* outM = outY + (size_t)BATCH * NTOK * DMODEL;

    size_t lds = (size_t)LDS_ELEMS * sizeof(bf16);  // 229,376 B < 320 KB/WGP

    mixattn_kernel<<<BATCH, 256, lds, stream>>>(
        x, y, nmask, WkvXT, WkvYT, WqxT, WqyT, WpT, WpyT,
        qkv_b, qkv_y_b, qx_b, qy_b, p_b, py_b, biasWH, outX, outY, outM);
}
